// GIBGIN_75960791597152
// MI455X (gfx1250) — matrix operand order repaired and checked by reading, not compiled, not run on hardware
//
#include <hip/hip_runtime.h>

// ---------------------------------------------------------------------------
// GIB-GIN forward for MI455X (gfx1250, wave32, WMMA).
//   B=512 graphs x 128 nodes, N=65536, E=262144 (512 edges/graph, contiguous
//   per graph), F_IN=128, H=256, C=2.
// GEMMs run as bf16 WMMA (v_wmma_f32_16x16x32_bf16) with full weight matrix
// staged in LDS; edge aggregation runs per-graph fully inside LDS.
// ---------------------------------------------------------------------------

typedef unsigned int u32;
typedef unsigned short u16;

constexpr int NGRAPH = 512;
constexpr int NPG    = 128;
constexpr int NNODE  = NGRAPH * NPG;   // 65536
constexpr int NEDGE  = 262144;
constexpr int EPG    = NEDGE / NGRAPH; // 512
constexpr int HDIM   = 256;
constexpr int FIN    = 128;
constexpr int NOUTC  = 256;            // all WMMA GEMMs have 256 output cols

enum { ACT_NONE = 0, ACT_RELU = 1, ACT_TANH = 2 };

typedef __attribute__((ext_vector_type(16))) __bf16 v16bf;
typedef __attribute__((ext_vector_type(8)))  float  v8f;

union BFrag { v16bf v; u32 u[8]; };

__device__ __forceinline__ u16 f2bf(float f) {
  u32 u = __float_as_uint(f);
  u += 0x7FFFu + ((u >> 16) & 1u);            // round-to-nearest-even
  return (u16)(u >> 16);
}
__device__ __forceinline__ float bf2f(u16 h) {
  return __uint_as_float(((u32)h) << 16);
}

// ---------------------------------------------------------------------------
// Weight convert: fp32 [K, Nout] row-major  ->  bf16 transposed [Nout, K]
// ---------------------------------------------------------------------------
__global__ void wconv_kernel(const float* __restrict__ w, u16* __restrict__ wt,
                             int K, int Nn) {
  int id = blockIdx.x * 256 + threadIdx.x;
  if (id >= K * Nn) return;
  int k = id / Nn, n = id - k * Nn;
  wt[n * K + k] = f2bf(w[id]);
}

// ---------------------------------------------------------------------------
// Per-graph GIN aggregation: z = h + sum_{(s,d) in E_g} h[s] scattered to d.
// Whole graph lives in LDS; thread t owns feature column t -> race-free,
// no atomics, no intra-loop barriers.
// ---------------------------------------------------------------------------
template <int F>
__global__ __launch_bounds__(256) void agg_kernel(const float* __restrict__ h,
                                                  const int* __restrict__ src,
                                                  const int* __restrict__ dst,
                                                  u16* __restrict__ zb) {
  extern __shared__ float fsm[];
  float* hl = fsm;                  // [128*F]
  float* zl = fsm + 128 * F;        // [128*F]
  int*   el = (int*)(fsm + 2 * 128 * F);  // [2*EPG]
  const int g = blockIdx.x, tid = threadIdx.x;
  const int nbase = g * NPG;
  for (int i = tid; i < 128 * F; i += 256) {
    float v = h[(size_t)nbase * F + i];
    hl[i] = v; zl[i] = v;
  }
  const int ebase = g * EPG;
  for (int i = tid; i < EPG; i += 256) {
    el[i]       = src[ebase + i] - nbase;
    el[EPG + i] = dst[ebase + i] - nbase;
  }
  __syncthreads();
  for (int e = 0; e < EPG; ++e) {
    int s = el[e], d = el[EPG + e];
    for (int f = tid; f < F; f += 256) zl[d * F + f] += hl[s * F + f];
  }
  __syncthreads();
  for (int i = tid; i < 128 * F; i += 256)
    zb[(size_t)nbase * F + i] = f2bf(zl[i]);
}

// ---------------------------------------------------------------------------
// WMMA GEMM: out[M,256] = act(A[M,K](bf16) @ W[K,256] + bias)
//   Wt: bf16 weights pre-transposed to [256,K].
//   Block = 256 threads = 8 waves; block computes 128 rows x 256 cols.
//   Full Wt + 128-row A tile staged in LDS; wave w owns rows w*16..w*16+15.
// ---------------------------------------------------------------------------
template <int K, int ACT, bool WF32, bool WBF>
__global__ __launch_bounds__(256) void gemm_kernel(const u16* __restrict__ A,
                                                   const u16* __restrict__ Wt,
                                                   const float* __restrict__ bias,
                                                   float* __restrict__ outF,
                                                   u16* __restrict__ outB) {
  extern __shared__ u16 smem[];
  u16* Wl = smem;                    // [NOUTC*K]  (transposed: [n][k])
  u16* Al = smem + NOUTC * K;        // [128*K]
  const int tid = threadIdx.x;

  __builtin_prefetch(A + (size_t)blockIdx.x * 128 * K, 0, 1);  // global_prefetch_b8
  // stage full weight matrix
  {
    const u32* s = (const u32*)Wt;
    u32* d = (u32*)Wl;
    constexpr int cnt = NOUTC * K / 2;
    for (int i = tid; i < cnt; i += 256) d[i] = s[i];
  }
  // stage 128-row A tile
  {
    const u32* s = (const u32*)(A + (size_t)blockIdx.x * 128 * K);
    u32* d = (u32*)Al;
    constexpr int cnt = 128 * K / 2;
    for (int i = tid; i < cnt; i += 256) d[i] = s[i];
  }
  __syncthreads();

  const int wave = tid >> 5;
  const int lane = tid & 31;
  const int hrow = lane & 15;   // A: row-in-tile, B/D: col-in-tile
  const int half = lane >> 4;
  constexpr int KT = K / 32;

  // Preload all A fragments for this wave's 16 rows (ISA 16-bit A layout:
  // VGPR v holds K = (v&3)*2 + (v>>2)*16 + half*8, +1).
  BFrag af[KT];
  const u16* Arow = Al + (wave * 16 + hrow) * K;
#pragma unroll
  for (int kt = 0; kt < KT; ++kt) {
#pragma unroll
    for (int v = 0; v < 8; ++v) {
      int k = kt * 32 + (v & 3) * 2 + ((v >> 2) * 16) + half * 8;
      af[kt].u[v] = *(const u32*)(Arow + k);
    }
  }

#pragma unroll 1
  for (int nt = 0; nt < NOUTC / 16; ++nt) {
    v8f acc = {};
    // B fragment: lane's column, same K order as A (table layout) in transposed-W LDS.
    const u16* Wrow = Wl + (nt * 16 + hrow) * K + half * 8;
#pragma unroll
    for (int kt = 0; kt < KT; ++kt) {
      BFrag bf;
#pragma unroll
      for (int v = 0; v < 8; ++v) bf.u[v] = *(const u32*)(Wrow + kt * 32 + (v & 3) * 2 + (v >> 2) * 16);
      acc = __builtin_amdgcn_wmma_f32_16x16x32_bf16(
          false, af[kt].v, false, bf.v, (short)0, acc, false, false);
    }
    const int col = nt * 16 + hrow;
    const float b = bias[col];
#pragma unroll
    for (int r = 0; r < 8; ++r) {
      const int m = blockIdx.x * 128 + wave * 16 + half * 8 + r;
      float v = acc[r] + b;
      if (ACT == ACT_RELU) v = v > 0.f ? v : 0.f;
      if (ACT == ACT_TANH) v = tanhf(v);
      const size_t o = (size_t)m * NOUTC + col;
      if (WF32) outF[o] = v;
      if (WBF)  outB[o] = f2bf(v);
    }
  }
}

// ---------------------------------------------------------------------------
// BatchNorm (training-mode batch stats), deterministic two-stage reduction.
// ---------------------------------------------------------------------------
__global__ __launch_bounds__(256) void bn_partial_kernel(const float* __restrict__ z,
                                                         float* __restrict__ ps,
                                                         float* __restrict__ pss) {
  const int col = threadIdx.x, b = blockIdx.x;
  float s = 0.f, ss = 0.f;
  const float* p = z + (size_t)b * 256 * HDIM;
  for (int r = 0; r < 256; ++r) { float v = p[r * HDIM + col]; s += v; ss += v * v; }
  ps[b * HDIM + col] = s; pss[b * HDIM + col] = ss;
}

__global__ __launch_bounds__(256) void bn_finalize_kernel(const float* __restrict__ ps,
                                                          const float* __restrict__ pss,
                                                          const float* __restrict__ gamma,
                                                          const float* __restrict__ beta,
                                                          float* __restrict__ scale,
                                                          float* __restrict__ shift) {
  const int c = threadIdx.x;
  float s = 0.f, ss = 0.f;
  for (int b = 0; b < 256; ++b) { s += ps[b * HDIM + c]; ss += pss[b * HDIM + c]; }
  const float inv = 1.f / (float)NNODE;
  float mu  = s * inv;
  float var = ss * inv - mu * mu;
  float sc  = gamma[c] * rsqrtf(var + 1e-5f);
  scale[c] = sc; shift[c] = beta[c] - mu * sc;
}

__global__ __launch_bounds__(256) void bn_apply_kernel(const float* __restrict__ z,
                                                       const float* __restrict__ scale,
                                                       const float* __restrict__ shift,
                                                       float* __restrict__ hf,
                                                       u16* __restrict__ hb) {
  const size_t i = (size_t)blockIdx.x * 256 + threadIdx.x;
  const int c = threadIdx.x;  // 256 cols, 256 threads -> col == tid
  float v = z[i] * scale[c] + shift[c];
  hf[i] = v;
  if (hb) hb[i] = f2bf(v);
}

// ---------------------------------------------------------------------------
// a = softmax(t @ c2w + c2b), t = tanh(...) precomputed (bf16), c2: 256->2
// ---------------------------------------------------------------------------
__global__ __launch_bounds__(256) void assign_kernel(const u16* __restrict__ t,
                                                     const float* __restrict__ c2w,
                                                     const float* __restrict__ c2b,
                                                     float* __restrict__ a) {
  const int i = blockIdx.x * 256 + threadIdx.x;
  const u16* tr = t + (size_t)i * HDIM;
  float s0 = c2b[0], s1 = c2b[1];
  for (int k = 0; k < HDIM; ++k) {
    float v = bf2f(tr[k]);
    s0 += v * c2w[2 * k]; s1 += v * c2w[2 * k + 1];
  }
  float m = fmaxf(s0, s1);
  float e0 = __expf(s0 - m), e1 = __expf(s1 - m);
  float inv = 1.f / (e0 + e1);
  a[2 * i] = e0 * inv; a[2 * i + 1] = e1 * inv;
}

// ---------------------------------------------------------------------------
// Per-graph pooling: sub = sum a0*h, graph = mean h. Thread = feature col.
// ---------------------------------------------------------------------------
__global__ __launch_bounds__(256) void pool_kernel(const float* __restrict__ a,
                                                   const float* __restrict__ h,
                                                   float* __restrict__ subOut,
                                                   u16* __restrict__ subBf,
                                                   float* __restrict__ graphOut) {
  const int g = blockIdx.x, f = threadIdx.x;
  float s = 0.f, gr = 0.f;
  const float* hp = h + (size_t)g * NPG * HDIM;
  const float* ap = a + (size_t)g * NPG * 2;
  for (int n = 0; n < NPG; ++n) {
    float hv = hp[n * HDIM + f];
    s += ap[2 * n] * hv; gr += hv;
  }
  subOut[g * HDIM + f]  = s;
  subBf[g * HDIM + f]   = f2bf(s);
  graphOut[g * HDIM + f] = gr * (1.f / (float)NPG);
}

// ---------------------------------------------------------------------------
// Pooled adjacency [2x2] per graph, row L1-normalize, diag MSE penalty.
// ---------------------------------------------------------------------------
__global__ __launch_bounds__(256) void adj_kernel(const float* __restrict__ a,
                                                  const int* __restrict__ src,
                                                  const int* __restrict__ dst,
                                                  float* __restrict__ pen) {
  __shared__ float r0[256], r1[256], r2[256], r3[256];
  const int g = blockIdx.x, tid = threadIdx.x;
  float m00 = 0.f, m01 = 0.f, m10 = 0.f, m11 = 0.f;
  const int eb = g * EPG;
  for (int e = tid; e < EPG; e += 256) {
    int s = src[eb + e], d = dst[eb + e];
    float as0 = a[2 * s], as1 = a[2 * s + 1];
    float ad0 = a[2 * d], ad1 = a[2 * d + 1];
    m00 += as0 * ad0; m01 += as0 * ad1; m10 += as1 * ad0; m11 += as1 * ad1;
  }
  r0[tid] = m00; r1[tid] = m01; r2[tid] = m10; r3[tid] = m11;
  __syncthreads();
  for (int st = 128; st > 0; st >>= 1) {
    if (tid < st) { r0[tid] += r0[tid + st]; r1[tid] += r1[tid + st];
                    r2[tid] += r2[tid + st]; r3[tid] += r3[tid + st]; }
    __syncthreads();
  }
  if (tid == 0) {
    float rn0 = fmaxf(fabsf(r0[0]) + fabsf(r1[0]), 1e-12f);
    float rn1 = fmaxf(fabsf(r2[0]) + fabsf(r3[0]), 1e-12f);
    float d0 = r0[0] / rn0 - 1.f, d1 = r3[0] / rn1 - 1.f;
    pen[g] = 0.5f * (d0 * d0 + d1 * d1);
  }
}

__global__ __launch_bounds__(256) void pen_reduce_kernel(const float* __restrict__ pen,
                                                         float* __restrict__ out) {
  __shared__ float s[256];
  const int tid = threadIdx.x;
  s[tid] = pen[tid] + pen[tid + 256];
  __syncthreads();
  for (int st = 128; st > 0; st >>= 1) {
    if (tid < st) s[tid] += s[tid + st];
    __syncthreads();
  }
  if (tid == 0) out[0] = s[0] * (1.f / (float)NGRAPH);
}

// ---------------------------------------------------------------------------
// Head: out = log_softmax(z @ l2w + l2b), z (bf16) [512,256], l2: 256->2
// ---------------------------------------------------------------------------
__global__ __launch_bounds__(256) void head_kernel(const u16* __restrict__ z,
                                                   const float* __restrict__ w,
                                                   const float* __restrict__ b,
                                                   float* __restrict__ out) {
  const int g = blockIdx.x * 256 + threadIdx.x;
  const u16* zr = z + (size_t)g * HDIM;
  float s0 = b[0], s1 = b[1];
  for (int k = 0; k < HDIM; ++k) {
    float v = bf2f(zr[k]);
    s0 += v * w[2 * k]; s1 += v * w[2 * k + 1];
  }
  float m = fmaxf(s0, s1);
  float l = logf(__expf(s0 - m) + __expf(s1 - m));
  out[2 * g] = s0 - m - l; out[2 * g + 1] = s1 - m - l;
}

// ---------------------------------------------------------------------------
// Host launch
// ---------------------------------------------------------------------------
extern "C" void kernel_launch(void* const* d_in, const int* in_sizes, int n_in,
                              void* d_out, int out_size, void* d_ws, size_t ws_size,
                              hipStream_t stream) {
  (void)in_sizes; (void)n_in; (void)out_size; (void)ws_size;
  const float* x    = (const float*)d_in[0];
  const int*   ei   = (const int*)d_in[1];
  // d_in[2] = batch (implicit from contiguous layout)
  const float* w0_1 = (const float*)d_in[3];
  const float* b0_1 = (const float*)d_in[4];
  const float* w0_2 = (const float*)d_in[5];
  const float* b0_2 = (const float*)d_in[6];
  const float* g0   = (const float*)d_in[7];
  const float* be0  = (const float*)d_in[8];
  const float* wl1  = (const float*)d_in[9];
  const float* bl1  = (const float*)d_in[10];
  const float* wl2  = (const float*)d_in[11];
  const float* bl2  = (const float*)d_in[12];
  const float* gl   = (const float*)d_in[13];
  const float* bel  = (const float*)d_in[14];
  const float* c1w  = (const float*)d_in[15];
  const float* c1b  = (const float*)d_in[16];
  const float* c2w  = (const float*)d_in[17];
  const float* c2b  = (const float*)d_in[18];
  const float* l1w  = (const float*)d_in[19];
  const float* l1b  = (const float*)d_in[20];
  const float* l2w  = (const float*)d_in[21];
  const float* l2b  = (const float*)d_in[22];
  const int* src = ei;
  const int* dst = ei + NEDGE;

  // Workspace layout (~203 MB total)
  char* ws = (char*)d_ws;
  size_t off = 0;
  auto take = [&](size_t bytes) -> char* {
    char* p = ws + off;
    off += (bytes + 255) & ~(size_t)255;
    return p;
  };
  float* h_f32 = (float*)take((size_t)NNODE * HDIM * 4);
  u16*   z_bf  = (u16*)  take((size_t)NNODE * HDIM * 2);
  u16*   t_bf  = (u16*)  take((size_t)NNODE * HDIM * 2);
  float* z2    = (float*)take((size_t)NNODE * HDIM * 4);
  u16* w01t  = (u16*)take((size_t)FIN * HDIM * 2);
  u16* w02t  = (u16*)take((size_t)HDIM * HDIM * 2);
  u16* wl1t0 = (u16*)take((size_t)HDIM * HDIM * 2);
  u16* wl1t1 = (u16*)take((size_t)HDIM * HDIM * 2);
  u16* wl2t0 = (u16*)take((size_t)HDIM * HDIM * 2);
  u16* wl2t1 = (u16*)take((size_t)HDIM * HDIM * 2);
  u16* c1wt  = (u16*)take((size_t)HDIM * HDIM * 2);
  u16* l1wt  = (u16*)take((size_t)HDIM * HDIM * 2);
  float* a_f32 = (float*)take((size_t)NNODE * 2 * 4);
  float* psum  = (float*)take((size_t)256 * HDIM * 4);
  float* psum2 = (float*)take((size_t)256 * HDIM * 4);
  float* scale = (float*)take(HDIM * 4);
  float* shift = (float*)take(HDIM * 4);
  float* pen   = (float*)take(NGRAPH * 4);
  u16* sub_bf  = (u16*)take((size_t)NGRAPH * HDIM * 2);
  u16* zh_bf   = (u16*)take((size_t)NGRAPH * HDIM * 2);

  float* outp      = (float*)d_out;
  float* out_logit = outp;                                   // [512,2]
  float* out_sub   = outp + 1024;                            // [512,256]
  float* out_graph = outp + 1024 + NGRAPH * HDIM;            // [512,256]
  float* out_pen   = outp + 1024 + 2 * NGRAPH * HDIM;        // [1]

  auto cdiv = [](int a, int b) { return (a + b - 1) / b; };

  // --- weight conversion (fp32 -> bf16 transposed) ---
  wconv_kernel<<<cdiv(FIN * HDIM, 256), 256, 0, stream>>>(w0_1, w01t, FIN, HDIM);
  wconv_kernel<<<cdiv(HDIM * HDIM, 256), 256, 0, stream>>>(w0_2, w02t, HDIM, HDIM);
  wconv_kernel<<<cdiv(HDIM * HDIM, 256), 256, 0, stream>>>(wl1, wl1t0, HDIM, HDIM);
  wconv_kernel<<<cdiv(HDIM * HDIM, 256), 256, 0, stream>>>(wl1 + HDIM * HDIM, wl1t1, HDIM, HDIM);
  wconv_kernel<<<cdiv(HDIM * HDIM, 256), 256, 0, stream>>>(wl2, wl2t0, HDIM, HDIM);
  wconv_kernel<<<cdiv(HDIM * HDIM, 256), 256, 0, stream>>>(wl2 + HDIM * HDIM, wl2t1, HDIM, HDIM);
  wconv_kernel<<<cdiv(HDIM * HDIM, 256), 256, 0, stream>>>(c1w, c1wt, HDIM, HDIM);
  wconv_kernel<<<cdiv(HDIM * HDIM, 256), 256, 0, stream>>>(l1w, l1wt, HDIM, HDIM);

  const size_t aggShm128  = (size_t)2 * 128 * FIN * 4 + 2 * EPG * 4;   // 135 KB
  const size_t aggShm256  = (size_t)2 * 128 * HDIM * 4 + 2 * EPG * 4;  // 266 KB
  const size_t gemmShm128 = (size_t)(NOUTC * FIN + 128 * FIN) * 2;     // 96 KB
  const size_t gemmShm256 = (size_t)(NOUTC * HDIM + 128 * HDIM) * 2;   // 192 KB
  const int MB = NNODE / 128;  // GEMM blocks over nodes

  // ---------------- GIN layer 0 (F_IN=128 -> H -> H) ----------------
  agg_kernel<FIN><<<NGRAPH, 256, aggShm128, stream>>>(x, src, dst, z_bf);
  gemm_kernel<FIN, ACT_RELU, false, true><<<MB, 256, gemmShm128, stream>>>(z_bf, w01t, b0_1, nullptr, t_bf);
  gemm_kernel<HDIM, ACT_RELU, true, false><<<MB, 256, gemmShm256, stream>>>(t_bf, w02t, b0_2, z2, nullptr);
  bn_partial_kernel<<<256, 256, 0, stream>>>(z2, psum, psum2);
  bn_finalize_kernel<<<1, 256, 0, stream>>>(psum, psum2, g0, be0, scale, shift);
  bn_apply_kernel<<<NNODE, 256, 0, stream>>>(z2, scale, shift, h_f32, nullptr);

  // ---------------- GIN layer 1 ----------------
  agg_kernel<HDIM><<<NGRAPH, 256, aggShm256, stream>>>(h_f32, src, dst, z_bf);
  gemm_kernel<HDIM, ACT_RELU, false, true><<<MB, 256, gemmShm256, stream>>>(z_bf, wl1t0, bl1, nullptr, t_bf);
  gemm_kernel<HDIM, ACT_RELU, true, false><<<MB, 256, gemmShm256, stream>>>(t_bf, wl2t0, bl2, z2, nullptr);
  bn_partial_kernel<<<256, 256, 0, stream>>>(z2, psum, psum2);
  bn_finalize_kernel<<<1, 256, 0, stream>>>(psum, psum2, gl, bel, scale, shift);
  bn_apply_kernel<<<NNODE, 256, 0, stream>>>(z2, scale, shift, h_f32, nullptr);

  // ---------------- GIN layer 2 ----------------
  agg_kernel<HDIM><<<NGRAPH, 256, aggShm256, stream>>>(h_f32, src, dst, z_bf);
  gemm_kernel<HDIM, ACT_RELU, false, true><<<MB, 256, gemmShm256, stream>>>(z_bf, wl1t1, bl1 + HDIM, nullptr, t_bf);
  gemm_kernel<HDIM, ACT_RELU, true, false><<<MB, 256, gemmShm256, stream>>>(t_bf, wl2t1, bl2 + HDIM, z2, nullptr);
  bn_partial_kernel<<<256, 256, 0, stream>>>(z2, psum, psum2);
  bn_finalize_kernel<<<1, 256, 0, stream>>>(psum, psum2, gl + HDIM, bel + HDIM, scale, shift);
  bn_apply_kernel<<<NNODE, 256, 0, stream>>>(z2, scale, shift, h_f32, z_bf);  // also bf16 h

  // ---------------- assignment: a = softmax(tanh(h@c1w+c1b)@c2w+c2b) ----------------
  gemm_kernel<HDIM, ACT_TANH, false, true><<<MB, 256, gemmShm256, stream>>>(z_bf, c1wt, c1b, nullptr, t_bf);
  assign_kernel<<<NNODE / 256, 256, 0, stream>>>(t_bf, c2w, c2b, a_f32);

  // ---------------- pooling + connectivity penalty ----------------
  pool_kernel<<<NGRAPH, 256, 0, stream>>>(a_f32, h_f32, out_sub, sub_bf, out_graph);
  adj_kernel<<<NGRAPH, 256, 0, stream>>>(a_f32, src, dst, pen);
  pen_reduce_kernel<<<1, 256, 0, stream>>>(pen, out_pen);

  // ---------------- classifier head ----------------
  gemm_kernel<HDIM, ACT_RELU, false, true><<<NGRAPH / 128, 256, gemmShm256, stream>>>(sub_bf, l1wt, l1b, nullptr, zh_bf);
  head_kernel<<<NGRAPH / 256, 256, 0, stream>>>(zh_bf, l2w, l2b, out_logit);
}
